// EncoderRNN_67851893342796
// MI455X (gfx1250) — matrix operand order repaired and checked by reading, not compiled, not run on hardware
//
#include <hip/hip_runtime.h>
#include <hip/hip_bf16.h>
#include <math.h>

// Bidirectional LSTM encoder for MI455X (gfx1250, wave32, WMMA).
// Strategy:
//   1. transpose+convert the 8 HxH fp32 weight mats into bf16, column-major
//      (so WMMA B-fragments are contiguous 32B loads). 134MB fp32 -> 67MB bf16,
//      fits easily in the 192MB L2, so the 30 sequential recurrent matvecs
//      re-read U from L2, not HBM.
//   2. gather embeddings -> bf16 xs[16][2048] (row 15 zero pad).
//   3. one WMMA GEMM (16x8192x2048) precomputes x-projections + bias for all
//      steps (shared by both scan directions - weights are shared).
//   4. per step: one WMMA GEMM with A = zero-padded bf16 [h_fwd; h_bwd; 0...]
//      over all 4 U mats (16x8192x2048, 2 live rows), activation fused; then
//      an elementwise cell update that also re-emits h in bf16 so the next
//      step's GEMM inner loop is pure global_load_b128 + v_wmma.

#define HD   2048
#define G4   8192   // 4 gates * H columns
#define NT   15     // timesteps

typedef __attribute__((ext_vector_type(16))) __bf16 v16bf;
typedef __attribute__((ext_vector_type(8)))  float  v8f;

__device__ __forceinline__ unsigned short f2bf_bits(float f) {
    union { float f; unsigned u; } v; v.f = f;
    unsigned u = v.u;
    unsigned r = u + 0x7FFFu + ((u >> 16) & 1u);   // round-to-nearest-even
    return (unsigned short)(r >> 16);
}
__device__ __forceinline__ __bf16 bits2bf(unsigned short s) {
    union { unsigned short s; __bf16 b; } v; v.s = s; return v.b;
}
__device__ __forceinline__ float sigmoidf_(float x) {
    return 1.0f / (1.0f + __expf(-x));
}

// ---- WMMA fragment loaders (contiguous k order, same in A and B) ------------------

// B 32x16 bf16 fragment from transposed matrix MT[col][k] (k contiguous).
// lanes 0-15: col = n0+lane, K = k0..k0+15 ; lanes 16-31: K = k0+16..k0+31.
__device__ __forceinline__ v16bf load_bfrag(const unsigned short* MT, int n0, int k0) {
    int l    = threadIdx.x & 31;
    int n    = n0 + (l & 15);
    int koff = k0 + ((l >> 4) << 4);
    return *(const v16bf*)(MT + (size_t)n * HD + koff);
}

// A 16x32 bf16 fragment from bf16 row-major mat[16][2048].
// lane<16: row M=lane, elems 0-15 = K0..15
// lane>=16: row M=lane-16, elems 0-15 = K16..31
__device__ __forceinline__ v16bf load_afrag(const unsigned short* mat, int k0) {
    int l  = threadIdx.x & 31;
    int r  = l & 15;
    int kb = k0 + ((l >> 4) ? 16 : 0);
    const unsigned short* p = mat + (size_t)r * HD;
    v16bf a;
#pragma unroll
    for (int e = 0; e < 8; ++e) {
        a[e]     = bits2bf(p[kb + e]);
        a[8 + e] = bits2bf(p[kb + 8 + e]);
    }
    return a;
}

// ---- kernels ---------------------------------------------------------------

// Transpose+convert one HxH fp32 matrix (blockIdx.z selects which) into the
// combined bf16 transposed buffer: dst[(g*H + j)*H + k] = src[k*H + j].
__global__ void transpose_convert_kernel(
    const float* w_f, const float* w_i, const float* w_o, const float* w_c,
    const float* u_f, const float* u_i, const float* u_o, const float* u_c,
    unsigned short* WT, unsigned short* UT) {
    __shared__ float tile[32][33];
    int m = blockIdx.z;                 // 0..7 : W f,i,o,c then U f,i,o,c
    int g = m & 3;
    const float* src;
    switch (g) {
        case 0:  src = (m < 4) ? w_f : u_f; break;
        case 1:  src = (m < 4) ? w_i : u_i; break;
        case 2:  src = (m < 4) ? w_o : u_o; break;
        default: src = (m < 4) ? w_c : u_c; break;
    }
    unsigned short* dst = (m < 4) ? WT : UT;
    int k0 = blockIdx.x * 32;
    int j0 = blockIdx.y * 32;
    int tx = threadIdx.x, ty = threadIdx.y;   // 32 x 8
#pragma unroll
    for (int i = 0; i < 4; ++i)
        tile[ty + i * 8][tx] = src[(size_t)(k0 + ty + i * 8) * HD + j0 + tx];
    __syncthreads();
#pragma unroll
    for (int i = 0; i < 4; ++i) {
        int j = j0 + ty + i * 8;
        dst[(size_t)(g * HD + j) * HD + k0 + tx] = f2bf_bits(tile[tx][ty + i * 8]);
    }
}

// Gather embeddings -> bf16 xs (row 15 zero); init padded-h (rows 0-1 live,
// rows 2-15 permanently zero) and f32 cell state.
__global__ void gather_init_kernel(const int* idx, const float* emb, const float* hidden,
                                   unsigned short* xs, unsigned short* hpad, float* cbuf) {
    int tid = blockIdx.x * blockDim.x + threadIdx.x;   // 16*2048 threads
    int t = tid >> 11, k = tid & (HD - 1);
    unsigned short v = 0;
    if (t < NT) v = f2bf_bits(emb[(size_t)idx[t] * HD + k]);
    xs[(size_t)t * HD + k] = v;
    hpad[(size_t)t * HD + k] = (t == 0) ? f2bf_bits(hidden[k]) : (unsigned short)0;
    if (t == 0) cbuf[k] = 0.f;          // forward cell state
    if (t == 1) cbuf[HD + k] = 0.f;     // backward cell state
}

// pre[t][col] = xs[t] @ Wall[:,col] + bias(col), col = g*2048 + j.
__global__ void precompute_kernel(const unsigned short* xs, const unsigned short* WT,
                                  const float* b1, const float* b2,
                                  const float* b3, const float* b4, float* pre) {
    int wave = (blockIdx.x * blockDim.x + threadIdx.x) >> 5;
    int n0 = wave << 4;                 // 16-column tile
    v8f acc;
#pragma unroll
    for (int e = 0; e < 8; ++e) acc[e] = 0.f;
    for (int k0 = 0; k0 < HD; k0 += 32) {
        v16bf a = load_afrag(xs, k0);
        v16bf b = load_bfrag(WT, n0, k0);
        __builtin_prefetch(WT + (size_t)(n0 + (threadIdx.x & 15)) * HD + k0 + 256, 0, 1);
        acc = __builtin_amdgcn_wmma_f32_16x16x32_bf16(false, a, false, b,
                                                      (short)0, acc, false, false);
    }
    int l = threadIdx.x & 31;
    int col = n0 + (l & 15);
    int g = col >> 11, j = col & (HD - 1);
    const float* bp = (g == 0) ? b1 : (g == 1) ? b2 : (g == 2) ? b3 : b4;
    float bias = bp[j];
    int rbase = (l < 16) ? 0 : 8;       // D layout: VGPR r = rows r / r+8
#pragma unroll
    for (int r = 0; r < 8; ++r) {
        int row = rbase + r;
        if (row < NT) pre[(size_t)row * G4 + col] = acc[r] + bias;
    }
}

// z[dir][col] = act( [h_f;h_b] @ Uall[:,col] + pre[row(dir)][col] )
__global__ void gate_gemm_kernel(const unsigned short* hpad, const unsigned short* UT,
                                 const float* pre, int t, float* zbuf) {
    int wave = (blockIdx.x * blockDim.x + threadIdx.x) >> 5;
    int n0 = wave << 4;
    v8f acc;
#pragma unroll
    for (int e = 0; e < 8; ++e) acc[e] = 0.f;
    for (int k0 = 0; k0 < HD; k0 += 32) {
        v16bf a = load_afrag(hpad, k0);     // rows 2-15 are zero -> pure loads
        v16bf b = load_bfrag(UT, n0, k0);
        __builtin_prefetch(UT + (size_t)(n0 + (threadIdx.x & 15)) * HD + k0 + 256, 0, 1);
        acc = __builtin_amdgcn_wmma_f32_16x16x32_bf16(false, a, false, b,
                                                      (short)0, acc, false, false);
    }
    int l = threadIdx.x & 31;
    if (l < 16) {                       // rows 0,1 of D live in acc[0],acc[1], lanes 0-15
        int col = n0 + l;
        int g = col >> 11;
        float zf = acc[0] + pre[(size_t)t * G4 + col];
        float zb = acc[1] + pre[(size_t)(NT - 1 - t) * G4 + col];
        if (g == 3) { zf = tanhf(zf);     zb = tanhf(zb); }
        else        { zf = sigmoidf_(zf); zb = sigmoidf_(zb); }
        zbuf[col]      = zf;
        zbuf[G4 + col] = zb;
    }
}

// c = c*F + I*G ; h = O*tanh(c) ; h -> bf16 hpad rows 0-1 ; scatter outputs.
__global__ void cell_update_kernel(const float* zbuf, unsigned short* hpad, float* cbuf,
                                   float* out, int t) {
    int tid = blockIdx.x * blockDim.x + threadIdx.x;   // 4096 threads
    int dir = tid >> 11, j = tid & (HD - 1);
    const float* z = zbuf + (size_t)dir * G4;
    float F  = z[j];
    float I  = z[HD + j];
    float O  = z[2 * HD + j];
    float Gv = z[3 * HD + j];
    float c = cbuf[(size_t)dir * HD + j] * F + I * Gv;
    cbuf[(size_t)dir * HD + j] = c;
    float h = O * tanhf(c);
    hpad[(size_t)dir * HD + j] = f2bf_bits(h);         // next step's A matrix
    if (dir == 0) {
        out[(size_t)t * 4096 + j] = h;                       // out_f[t]
    } else {
        out[(size_t)(NT - 1 - t) * 4096 + 2048 + j] = h;     // out_b (reversed)
        if (t == NT - 1) out[(size_t)NT * 4096 + j] = h;     // final h_b
    }
}

// ---- launcher --------------------------------------------------------------

extern "C" void kernel_launch(void* const* d_in, const int* in_sizes, int n_in,
                              void* d_out, int out_size, void* d_ws, size_t ws_size,
                              hipStream_t stream) {
    const int*   input_batch = (const int*)d_in[0];
    const float* hidden = (const float*)d_in[1];
    const float* emb    = (const float*)d_in[2];
    const float* w_i = (const float*)d_in[3];
    const float* w_f = (const float*)d_in[4];
    const float* w_o = (const float*)d_in[5];
    const float* w_c = (const float*)d_in[6];
    const float* u_i = (const float*)d_in[7];
    const float* u_f = (const float*)d_in[8];
    const float* u_o = (const float*)d_in[9];
    const float* u_c = (const float*)d_in[10];
    const float* b1  = (const float*)d_in[11];
    const float* b2  = (const float*)d_in[12];
    const float* b3  = (const float*)d_in[13];
    const float* b4  = (const float*)d_in[14];
    float* out = (float*)d_out;

    // workspace carve (~68 MB)
    char* base = (char*)d_ws;
    size_t off = 0;
    auto carve = [&](size_t bytes) { char* p = base + off; off += (bytes + 255) & ~(size_t)255; return p; };
    unsigned short* WT   = (unsigned short*)carve((size_t)G4 * HD * 2);
    unsigned short* UT   = (unsigned short*)carve((size_t)G4 * HD * 2);
    unsigned short* xs   = (unsigned short*)carve((size_t)16 * HD * 2);
    unsigned short* hpad = (unsigned short*)carve((size_t)16 * HD * 2);
    float* pre  = (float*)carve((size_t)NT * G4 * 4);
    float* zbuf = (float*)carve((size_t)2 * G4 * 4);
    float* cbuf = (float*)carve((size_t)2 * HD * 4);

    transpose_convert_kernel<<<dim3(HD / 32, HD / 32, 8), dim3(32, 8), 0, stream>>>(
        w_f, w_i, w_o, w_c, u_f, u_i, u_o, u_c, WT, UT);

    gather_init_kernel<<<(16 * HD) / 256, 256, 0, stream>>>(
        input_batch, emb, hidden, xs, hpad, cbuf);

    precompute_kernel<<<G4 / 16 / 8, 256, 0, stream>>>(xs, WT, b1, b2, b3, b4, pre);

    for (int t = 0; t < NT; ++t) {
        gate_gemm_kernel<<<G4 / 16 / 8, 256, 0, stream>>>(hpad, UT, pre, t, zbuf);
        cell_update_kernel<<<(2 * HD) / 256, 256, 0, stream>>>(zbuf, hpad, cbuf, out, t);
    }
}
